// BondMessagePassing_18219251269756
// MI455X (gfx1250) — matrix operand order repaired and checked by reading, not compiled, not run on hardware
//
#include <hip/hip_runtime.h>

typedef unsigned short u16;
typedef __attribute__((ext_vector_type(16))) __bf16 v16bf;
typedef __attribute__((ext_vector_type(8)))  float  v8f;

__device__ __forceinline__ u16 f2bf(float f) {
  __bf16 h = (__bf16)f;
  return __builtin_bit_cast(u16, h);
}
__device__ __forceinline__ float bf2f(u16 u) {
  __bf16 h = __builtin_bit_cast(__bf16, u);
  return (float)h;
}

// ---- WMMA fragment loaders (bf16, 16x16x32) per CDNA5 ISA VGPR layouts ----
// A (16x32): lane group g = lane/16; VGPR 0..3 hold K = ks*32 + g*8 + 0..7,
//            VGPR 4..7 hold K = ks*32 + 16 + g*8 + 0..7  (row = lane%16)
__device__ __forceinline__ v16bf load_frag_a(const u16* lds, int row, int stride,
                                             int ks, int g) {
  const u16* p = lds + row * stride + ks * 32 + g * 8;
  union { uint4 u[2]; v16bf v; } f;
  f.u[0] = *(const uint4*)(p);
  f.u[1] = *(const uint4*)(p + 16);
  return f.v;
}
// B (32x16): lane holds column n = lane%16; group g holds K = ks*32 + g*8 + 0..7, ks*32 + 16 + g*8 + 0..7
// (weights stored row-major [n][k] in LDS => two runs of 8 consecutive halves, 16 apart)
__device__ __forceinline__ v16bf load_frag_b(const u16* lds, int col, int stride,
                                             int ks, int g) {
  const u16* p = lds + col * stride + ks * 32 + g * 8;
  union { uint4 u[2]; v16bf v; } f;
  f.u[0] = *(const uint4*)(p);
  f.u[1] = *(const uint4*)(p + 16);
  return f.v;
}

// One wave computes a 16-row x 128-col strip: 8 n-blocks, KSTEPS k-steps of 32.
template <int KSTEPS>
__device__ __forceinline__ void wave_gemm(const u16* Alds, int strideA,
                                          const u16* Blds, int strideB,
                                          int lane, int rowBase, v8f acc[8]) {
  const int r = rowBase + (lane & 15);
  const int g = lane >> 4;
#pragma unroll
  for (int ks = 0; ks < KSTEPS; ++ks) {
    v16bf a = load_frag_a(Alds, r, strideA, ks, g);
#pragma unroll
    for (int nb = 0; nb < 8; ++nb) {
      v16bf b = load_frag_b(Blds, nb * 16 + (lane & 15), strideB, ks, g);
      acc[nb] = __builtin_amdgcn_wmma_f32_16x16x32_bf16(
          false, a, false, b, (short)0, acc[nb], false, false);
    }
  }
}

// =====================  Kernel 1: edge init  =====================
// H0 = [x[src] || edge_attr] @ W_i^T + b_i ;  H = relu(H0) (bf16)
__global__ __launch_bounds__(256) void k_edge_init(
    const float* __restrict__ x, const float* __restrict__ ea,
    const float* __restrict__ Wi, const float* __restrict__ bi,
    const int* __restrict__ src, float* __restrict__ H0,
    u16* __restrict__ H, int E) {
  constexpr int KP = 160;  // K = 144 padded to 5 k-steps of 32
  __shared__ u16 Alds[128 * KP];
  __shared__ u16 Blds[128 * KP];
  __shared__ float bias[128];
  __shared__ int ssrc[128];
  const int tid = threadIdx.x;
  const int e0 = blockIdx.x * 128;

  if (tid < 128) {
    bias[tid] = bi[tid];
    int e = e0 + tid;
    ssrc[tid] = (e < E) ? src[e] : 0;
  }
  __syncthreads();

  // W_i row-major [128][144] -> bf16 LDS, zero-padded to 160
  for (int i = tid; i < 128 * (KP / 4); i += 256) {
    int row = i / (KP / 4);
    int k = (i % (KP / 4)) * 4;
    float4 w = make_float4(0.f, 0.f, 0.f, 0.f);
    if (k < 144) w = *(const float4*)(Wi + row * 144 + k);
    u16* q = Blds + row * KP + k;
    q[0] = f2bf(w.x); q[1] = f2bf(w.y); q[2] = f2bf(w.z); q[3] = f2bf(w.w);
  }
  // A tile: concat(x[src], edge_attr, zero pad)
  for (int i = tid; i < 128 * (KP / 4); i += 256) {
    int row = i / (KP / 4);
    int k = (i % (KP / 4)) * 4;
    int e = e0 + row;
    float4 v = make_float4(0.f, 0.f, 0.f, 0.f);
    if (e < E) {
      if (k < 128)      v = *(const float4*)(x + (long long)ssrc[row] * 128 + k);
      else if (k < 144) v = *(const float4*)(ea + (long long)e * 16 + (k - 128));
    }
    u16* q = Alds + row * KP + k;
    q[0] = f2bf(v.x); q[1] = f2bf(v.y); q[2] = f2bf(v.z); q[3] = f2bf(v.w);
  }
  __syncthreads();

  const int lane = tid & 31, wave = tid >> 5;
  v8f acc[8];
#pragma unroll
  for (int i = 0; i < 8; ++i) acc[i] = (v8f){0.f, 0.f, 0.f, 0.f, 0.f, 0.f, 0.f, 0.f};
  wave_gemm<5>(Alds, KP, Blds, KP, lane, wave * 16, acc);

  const int g = lane >> 4, cn = lane & 15;
#pragma unroll
  for (int nb = 0; nb < 8; ++nb) {
#pragma unroll
    for (int v = 0; v < 8; ++v) {
      int row = wave * 16 + g * 8 + v;
      int e = e0 + row;
      if (e < E) {
        int col = nb * 16 + cn;
        float val = acc[nb][v] + bias[col];
        H0[(long long)e * 128 + col] = val;
        H[(long long)e * 128 + col] = f2bf(fmaxf(val, 0.f));
      }
    }
  }
}

// =====================  Kernel: zero f32 buffer (float4)  =====================
__global__ __launch_bounds__(256) void k_zero(float* __restrict__ p, long long n4) {
  long long i = (long long)blockIdx.x * 256 + threadIdx.x;
  if (i < n4) ((float4*)p)[i] = make_float4(0.f, 0.f, 0.f, 0.f);
}

// =====================  Kernel: scatter-add H(bf16) into agg[node] (f32) ======
__global__ __launch_bounds__(256) void k_scatter(const u16* __restrict__ H,
                                                 const int* __restrict__ dst,
                                                 float* __restrict__ agg, int E) {
  long long idx = (long long)blockIdx.x * 256 + threadIdx.x;
  long long tot = (long long)E * 32;
  if (idx >= tot) return;
  int e = (int)(idx >> 5);
  int j = (int)(idx & 31) * 4;
  int d = dst[e];
  uint2 hv = *(const uint2*)(H + (long long)e * 128 + j);
  float* a = agg + (long long)d * 128 + j;
  atomicAdd(a + 0, bf2f((u16)(hv.x & 0xffffu)));
  atomicAdd(a + 1, bf2f((u16)(hv.x >> 16)));
  atomicAdd(a + 2, bf2f((u16)(hv.y & 0xffffu)));
  atomicAdd(a + 3, bf2f((u16)(hv.y >> 16)));
}

// =====================  Kernel: message + update GEMM  =====================
// M = agg[src] - H[rev];  Hout = relu(H0 + M @ W_h^T + b_h)
__global__ __launch_bounds__(256) void k_msg_gemm(
    const float* __restrict__ agg, const u16* __restrict__ Hin,
    const float* __restrict__ H0, const float* __restrict__ Wh,
    const float* __restrict__ bh, const int* __restrict__ src,
    const int* __restrict__ rev, u16* __restrict__ Hout, int E) {
  constexpr int KP = 136;  // 128 + pad to dodge LDS bank conflicts
  __shared__ u16 Alds[128 * KP];
  __shared__ u16 Blds[128 * KP];
  __shared__ float bias[128];
  __shared__ int ssrc[128];
  __shared__ int srev[128];
  const int tid = threadIdx.x;
  const int e0 = blockIdx.x * 128;

  if (tid < 128) {
    bias[tid] = bh[tid];
    int e = e0 + tid;
    ssrc[tid] = (e < E) ? src[e] : 0;
    srev[tid] = (e < E) ? rev[e] : -1;
  }
  __syncthreads();

  for (int i = tid; i < 128 * 32; i += 256) {  // W_h [128][128] -> bf16
    int row = i >> 5;
    int k = (i & 31) * 4;
    float4 w = *(const float4*)(Wh + row * 128 + k);
    u16* q = Blds + row * KP + k;
    q[0] = f2bf(w.x); q[1] = f2bf(w.y); q[2] = f2bf(w.z); q[3] = f2bf(w.w);
  }
  for (int i = tid; i < 128 * 32; i += 256) {  // M tile
    int row = i >> 5;
    int k = (i & 31) * 4;
    int e = e0 + row;
    float m0 = 0.f, m1 = 0.f, m2 = 0.f, m3 = 0.f;
    if (e < E) {
      float4 av = *(const float4*)(agg + (long long)ssrc[row] * 128 + k);
      float h0 = 0.f, h1 = 0.f, h2 = 0.f, h3 = 0.f;
      int r = srev[row];
      if (r >= 0) {
        uint2 hv = *(const uint2*)(Hin + (long long)r * 128 + k);
        h0 = bf2f((u16)(hv.x & 0xffffu)); h1 = bf2f((u16)(hv.x >> 16));
        h2 = bf2f((u16)(hv.y & 0xffffu)); h3 = bf2f((u16)(hv.y >> 16));
      }
      m0 = av.x - h0; m1 = av.y - h1; m2 = av.z - h2; m3 = av.w - h3;
    }
    u16* q = Alds + row * KP + k;
    q[0] = f2bf(m0); q[1] = f2bf(m1); q[2] = f2bf(m2); q[3] = f2bf(m3);
  }
  __syncthreads();

  const int lane = tid & 31, wave = tid >> 5;
  v8f acc[8];
#pragma unroll
  for (int i = 0; i < 8; ++i) acc[i] = (v8f){0.f, 0.f, 0.f, 0.f, 0.f, 0.f, 0.f, 0.f};
  wave_gemm<4>(Alds, KP, Blds, KP, lane, wave * 16, acc);

  const int g = lane >> 4, cn = lane & 15;
#pragma unroll
  for (int nb = 0; nb < 8; ++nb) {
#pragma unroll
    for (int v = 0; v < 8; ++v) {
      int row = wave * 16 + g * 8 + v;
      int e = e0 + row;
      if (e < E) {
        int col = nb * 16 + cn;
        float val = acc[nb][v] + bias[col] + H0[(long long)e * 128 + col];
        Hout[(long long)e * 128 + col] = f2bf(fmaxf(val, 0.f));
      }
    }
  }
}

// =====================  Kernel: node readout  =====================
// out = relu([x || M_node] @ W_o^T + b_o), isolated nodes use x as M_node
__global__ __launch_bounds__(256) void k_node_out(
    const float* __restrict__ x, const float* __restrict__ agg,
    const float* __restrict__ Wo, const float* __restrict__ bo,
    float* __restrict__ out, int N) {
  constexpr int KP = 264;  // 256 + pad
  __shared__ u16 Alds[128 * KP];
  __shared__ u16 Blds[128 * KP];
  __shared__ float bias[128];
  __shared__ float rowsum[128];
  const int tid = threadIdx.x;
  const int n0 = blockIdx.x * 128;

  if (tid < 128) {
    bias[tid] = bo[tid];
    rowsum[tid] = 0.f;
  }
  __syncthreads();

  for (int i = tid; i < 128 * 64; i += 256) {  // W_o [128][256] -> bf16
    int row = i >> 6;
    int k = (i & 63) * 4;
    float4 w = *(const float4*)(Wo + row * 256 + k);
    u16* q = Blds + row * KP + k;
    q[0] = f2bf(w.x); q[1] = f2bf(w.y); q[2] = f2bf(w.z); q[3] = f2bf(w.w);
  }
  for (int i = tid; i < 128 * 32; i += 256) {  // x part (cols 0..127)
    int row = i >> 5;
    int k = (i & 31) * 4;
    int n = n0 + row;
    float4 v = make_float4(0.f, 0.f, 0.f, 0.f);
    if (n < N) v = *(const float4*)(x + (long long)n * 128 + k);
    u16* q = Alds + row * KP + k;
    q[0] = f2bf(v.x); q[1] = f2bf(v.y); q[2] = f2bf(v.z); q[3] = f2bf(v.w);
  }
  for (int i = tid; i < 128 * 32; i += 256) {  // M_node part (cols 128..255) + rowsum
    int row = i >> 5;
    int k = (i & 31) * 4;
    int n = n0 + row;
    float4 v = make_float4(0.f, 0.f, 0.f, 0.f);
    if (n < N) v = *(const float4*)(agg + (long long)n * 128 + k);
    u16* q = Alds + row * KP + 128 + k;
    q[0] = f2bf(v.x); q[1] = f2bf(v.y); q[2] = f2bf(v.z); q[3] = f2bf(v.w);
    float s = v.x + v.y + v.z + v.w;
    if (s != 0.f) atomicAdd(&rowsum[row], s);
  }
  __syncthreads();
  for (int i = tid; i < 128 * 32; i += 256) {  // isolated fix: M_node := x
    int row = i >> 5;
    int k = (i & 31) * 4;
    if (rowsum[row] == 0.f) {
      u16* q = Alds + row * KP;
      q[128 + k + 0] = q[k + 0];
      q[128 + k + 1] = q[k + 1];
      q[128 + k + 2] = q[k + 2];
      q[128 + k + 3] = q[k + 3];
    }
  }
  __syncthreads();

  const int lane = tid & 31, wave = tid >> 5;
  v8f acc[8];
#pragma unroll
  for (int i = 0; i < 8; ++i) acc[i] = (v8f){0.f, 0.f, 0.f, 0.f, 0.f, 0.f, 0.f, 0.f};
  wave_gemm<8>(Alds, KP, Blds, KP, lane, wave * 16, acc);

  const int g = lane >> 4, cn = lane & 15;
#pragma unroll
  for (int nb = 0; nb < 8; ++nb) {
#pragma unroll
    for (int v = 0; v < 8; ++v) {
      int row = wave * 16 + g * 8 + v;
      int n = n0 + row;
      if (n < N) {
        int col = nb * 16 + cn;
        float val = acc[nb][v] + bias[col];
        out[(long long)n * 128 + col] = fmaxf(val, 0.f);
      }
    }
  }
}

// =====================  Host launcher  =====================
extern "C" void kernel_launch(void* const* d_in, const int* in_sizes, int n_in,
                              void* d_out, int out_size, void* d_ws, size_t ws_size,
                              hipStream_t stream) {
  const float* x  = (const float*)d_in[0];
  const float* ea = (const float*)d_in[1];
  const float* Wi = (const float*)d_in[2];
  const float* bi = (const float*)d_in[3];
  const float* Wh = (const float*)d_in[4];
  const float* bh = (const float*)d_in[5];
  const float* Wo = (const float*)d_in[6];
  const float* bo = (const float*)d_in[7];
  const int* eidx = (const int*)d_in[8];
  const int* rev  = (const int*)d_in[9];

  const int E = in_sizes[9];        // rev_edge_index has E elements
  const int N = in_sizes[0] / 128;  // x is [N,128]
  const int* src = eidx;            // edge_index[0]
  const int* dst = eidx + E;        // edge_index[1]

  char* ws = (char*)d_ws;
  size_t off = 0;
  auto alloc = [&](size_t bytes) -> char* {
    char* p = ws + off;
    off += (bytes + 255) & ~(size_t)255;
    return p;
  };
  float* H0  = (float*)alloc((size_t)E * 128 * sizeof(float));
  u16*   Ha  = (u16*)  alloc((size_t)E * 128 * sizeof(u16));
  u16*   Hb  = (u16*)  alloc((size_t)E * 128 * sizeof(u16));
  float* agg = (float*)alloc((size_t)N * 128 * sizeof(float));

  const int TE = (E + 127) / 128;
  const int TN = (N + 127) / 128;
  const long long aggN4 = (long long)N * 32;  // float4 count
  const int zgrid = (int)((aggN4 + 255) / 256);
  const long long sc = (long long)E * 32;
  const int sgrid = (int)((sc + 255) / 256);
  dim3 blk(256);

  k_edge_init<<<TE, blk, 0, stream>>>(x, ea, Wi, bi, src, H0, Ha, E);

  u16* cur = Ha;
  u16* nxt = Hb;
  for (int it = 1; it < 5; ++it) {  // DEPTH-1 message passes
    k_zero<<<zgrid, blk, 0, stream>>>(agg, aggN4);
    k_scatter<<<sgrid, blk, 0, stream>>>(cur, dst, agg, E);
    k_msg_gemm<<<TE, blk, 0, stream>>>(agg, cur, H0, Wh, bh, src, rev, nxt, E);
    u16* t = cur; cur = nxt; nxt = t;
  }
  k_zero<<<zgrid, blk, 0, stream>>>(agg, aggN4);
  k_scatter<<<sgrid, blk, 0, stream>>>(cur, dst, agg, E);
  k_node_out<<<TN, blk, 0, stream>>>(x, agg, Wo, bo, (float*)d_out, N);
}
